// SAModule_721554506012
// MI455X (gfx1250) — matrix operand order repaired and checked by reading, not compiled, not run on hardware
//
#include <hip/hip_runtime.h>
#include <hip/hip_bf16.h>

#define B_   8
#define N_   4096
#define M_   2048
#define F_   64
#define H1_  64
#define H2_  128
#define K_   32
#define KP_  96          // GEMM1 K padded 68 -> 96 (zero-padded weight rows)
#define R2_  0.04f

typedef __attribute__((ext_vector_type(16))) _Float16 v16h;
typedef __attribute__((ext_vector_type(8)))  _Float16 v8h;
typedef __attribute__((ext_vector_type(8)))  float    v8f;

union V16U { v16h v; v8h h[2]; };

// 16-bit A-matrix 16x32 fragment from row-major LDS (stride in halfs).
// ISA layout: lanes 0-15 row=M, K=k0+[0..7] (VGPR0-3) and k0+[16..23] (VGPR4-7);
// lanes 16-31 same rows, K offset +8.
__device__ __forceinline__ v16h load_fragA(const _Float16* base, int row, int stride,
                                           int k0, int lane) {
  const _Float16* p = base + (size_t)(row + (lane & 15)) * stride
                           + (k0 + ((lane & 16) ? 8 : 0));
  V16U u;
  u.h[0] = *(const v8h*)p;
  u.h[1] = *(const v8h*)(p + 16);
  return u.v;
}

// 16-bit B-matrix 32x16 fragment; weights stored N-major (transposed) in LDS so
// each lane's 16 K-values for its column n = col + (lane&15) form two contiguous runs of 8.
// lanes 0-15: K = k0+[0..7], k0+[16..23] ; lanes 16-31: K offset +8.
__device__ __forceinline__ v16h load_fragB(const _Float16* wt, int col, int stride,
                                           int k0, int lane) {
  const _Float16* p = wt + (size_t)(col + (lane & 15)) * stride
                         + (k0 + ((lane & 16) ? 8 : 0));
  V16U u;
  u.h[0] = *(const v8h*)p;
  u.h[1] = *(const v8h*)(p + 16);
  return u.v;
}

__device__ __forceinline__ float get_angle(float ax, float ay, float az,
                                           float bx, float by, float bz) {
  float cx = ay*bz - az*by;
  float cy = az*bx - ax*bz;
  float cz = ax*by - ay*bx;
  float cn2 = cx*cx + cy*cy + cz*cz;
  float cn = cn2 > 0.f ? sqrtf(cn2) : 0.f;
  float d  = ax*bx + ay*by + az*bz;
  if (cn == 0.f && d == 0.f) d = 1.f;   // torch atan2(0,0)==0 match
  return atan2f(cn, d);
}

// ---------------- Kernel 1: farthest point sampling, one WG per batch -------
__global__ void __launch_bounds__(1024)
fps_kernel(const float* __restrict__ pos, int* __restrict__ idxOut) {
  extern __shared__ char smem[];
  float* px   = (float*)smem;
  float* py   = px + N_;
  float* pz   = py + N_;
  float* redV = pz + N_;
  int*   redI = (int*)(redV + 32);
  int*   win  = redI + 32;

  const int b = blockIdx.x;
  const int tid = threadIdx.x;
  const int lane = tid & 31, wid = tid >> 5;
  const float* pb = pos + (size_t)b * N_ * 3;

  for (int i = tid; i < N_; i += 1024) {
    px[i] = pb[i*3+0]; py[i] = pb[i*3+1]; pz[i] = pb[i*3+2];
  }
  if (tid == 0) { idxOut[(size_t)b*M_] = 0; win[0] = 0; }
  __syncthreads();

  float dst[4];
  #pragma unroll
  for (int q = 0; q < 4; ++q) dst[q] = 3.4e38f;

  int last = 0;
  for (int m = 1; m < M_; ++m) {
    const float lx = px[last], ly = py[last], lz = pz[last];
    float bv = -1.0f; int bi = 0x7fffffff;
    #pragma unroll
    for (int q = 0; q < 4; ++q) {
      const int p = q*1024 + tid;
      float dx = px[p]-lx, dy = py[p]-ly, dz = pz[p]-lz;
      float d = dx*dx + dy*dy + dz*dz;
      float nd = fminf(dst[q], d);
      dst[q] = nd;
      if (nd > bv) { bv = nd; bi = p; }     // ascending p: first max kept
    }
    #pragma unroll
    for (int off = 16; off > 0; off >>= 1) {
      float ov = __shfl_xor(bv, off, 32);
      int   oi = __shfl_xor(bi, off, 32);
      if (ov > bv || (ov == bv && oi < bi)) { bv = ov; bi = oi; }
    }
    if (lane == 0) { redV[wid] = bv; redI[wid] = bi; }
    __syncthreads();
    if (wid == 0) {
      bv = redV[lane]; bi = redI[lane];
      #pragma unroll
      for (int off = 16; off > 0; off >>= 1) {
        float ov = __shfl_xor(bv, off, 32);
        int   oi = __shfl_xor(bi, off, 32);
        if (ov > bv || (ov == bv && oi < bi)) { bv = ov; bi = oi; }
      }
      if (lane == 0) { win[0] = bi; idxOut[(size_t)b*M_ + m] = bi; }
    }
    __syncthreads();
    last = win[0];
  }
}

// ---------------- Kernel 2: K=32 nearest (ascending d2), one wave/center ----
__global__ void __launch_bounds__(256)
topk_kernel(const float* __restrict__ pos, const int* __restrict__ idxArr,
            int* __restrict__ nbrOut) {
  extern __shared__ char smem[];
  float* dbuf = (float*)smem;              // 8 waves * 4096
  const int tid = threadIdx.x, lane = tid & 31, wid = tid >> 5;
  const int g = blockIdx.x * 8 + wid;      // global center id
  const int b = g >> 11, m = g & (M_ - 1);
  const int ci = idxArr[(size_t)b*M_ + m];
  const float* pb = pos + (size_t)b * N_ * 3;
  const float cx = pb[ci*3+0], cy = pb[ci*3+1], cz = pb[ci*3+2];
  float* dl = dbuf + wid * N_;

  for (int t = lane; t < N_; t += 32) {
    float dx = pb[t*3+0]-cx, dy = pb[t*3+1]-cy, dz = pb[t*3+2]-cz;
    dl[t] = dx*dx + dy*dy + dz*dz;
  }
  // wave-local LDS: per-wave DS ordering, no barrier needed

  unsigned sel[4] = {0u,0u,0u,0u};         // 128 strip slots per lane
  for (int r = 0; r < K_; ++r) {
    float bv = 3.4e38f; int bi = 0x7fffffff;
    for (int i = 0; i < 128; ++i) {
      if (!((sel[i >> 5] >> (i & 31)) & 1u)) {
        const int p = (i << 5) + lane;     // ascending p within lane
        float v = dl[p];
        if (v < bv) { bv = v; bi = p; }
      }
    }
    #pragma unroll
    for (int off = 16; off > 0; off >>= 1) {
      float ov = __shfl_xor(bv, off, 32);
      int   oi = __shfl_xor(bi, off, 32);
      if (ov < bv || (ov == bv && oi < bi)) { bv = ov; bi = oi; }
    }
    if (lane == 0) nbrOut[(size_t)g * K_ + r] = bi;
    if ((bi & 31) == lane) {
      const int i = bi >> 5;
      sel[i >> 5] |= (1u << (i & 31));
    }
  }
}

// ---------------- Kernel 3: gather + PPF + 2x WMMA GEMM + masked max --------
// WG = 256 threads (8 waves) = 8 centers * 32 neighbors = 256 msg rows.
// Wave w owns center w of the tile: rows w*32 .. w*32+31.
__global__ void __launch_bounds__(256)
ppf_mlp_kernel(const float* __restrict__ x, const float* __restrict__ pos,
               const float* __restrict__ norm,
               const float* __restrict__ W1, const float* __restrict__ b1,
               const float* __restrict__ W2, const float* __restrict__ b2,
               const int* __restrict__ idxArr, const int* __restrict__ nbrArr,
               float* __restrict__ out, float* __restrict__ outPos,
               float* __restrict__ outNorm) {
  extern __shared__ char smem[];
  _Float16* msgL = (_Float16*)smem;        // 256 x 96 (row-major, f16)
  _Float16* wt1  = msgL + 256*KP_;         // 64 x 96  (N-major = W1^T, zero-padded K)
  _Float16* wt2  = wt1  + H1_*KP_;         // 128 x 64 (N-major = W2^T)
  _Float16* hL   = wt2  + H2_*H1_;         // 256 x 64 hidden, f16
  float* b1L   = (float*)(hL + 256*H1_);
  float* b2L   = b1L + H1_;
  float* flagL = b2L + H2_;                // 256 per-row valid flags

  const int tid = threadIdx.x, lane = tid & 31, wid = tid >> 5;
  const int g = blockIdx.x * 8 + wid;      // this wave's center
  const int b = g >> 11, m = g & (M_ - 1);

  // ---- stage weights (transposed, f16) + biases ----
  for (int i = tid; i < H1_*KP_; i += 256) {
    const int n = i / KP_, k = i % KP_;
    wt1[i] = (k < F_ + 4) ? (_Float16)W1[k*H1_ + n] : (_Float16)0.f;
  }
  for (int i = tid; i < H2_*H1_; i += 256) {
    const int n = i / H1_, k = i % H1_;
    wt2[i] = (_Float16)W2[k*H2_ + n];
  }
  if (tid < H1_) b1L[tid] = b1[tid];
  if (tid < H2_) b2L[tid] = b2[tid];

  // ---- gather one msg row per thread: row tid = (center wid, neighbor lane) ----
  {
    const int ci = idxArr[(size_t)b*M_ + m];
    const int j  = nbrArr[(size_t)g*K_ + lane];
    const float* pb = pos  + (size_t)b*N_*3;
    const float* nb = norm + (size_t)b*N_*3;
    const float pix = pb[ci*3+0], piy = pb[ci*3+1], piz = pb[ci*3+2];
    const float nix = nb[ci*3+0], niy = nb[ci*3+1], niz = nb[ci*3+2];
    const float pjx = pb[j*3+0],  pjy = pb[j*3+1],  pjz = pb[j*3+2];
    const float njx = nb[j*3+0],  njy = nb[j*3+1],  njz = nb[j*3+2];
    const float dx = pjx-pix, dy = pjy-piy, dz = pjz-piz;
    const float d2 = dx*dx + dy*dy + dz*dz;
    const float p0 = d2 > 0.f ? sqrtf(d2) : 0.f;
    const float an1 = get_angle(nix,niy,niz, dx,dy,dz);
    const float an2 = get_angle(njx,njy,njz, dx,dy,dz);
    const float an3 = get_angle(nix,niy,niz, njx,njy,njz);

    _Float16* row = msgL + tid * KP_;
    const float* xr = x + ((size_t)b*N_ + j) * F_;
    __builtin_prefetch(xr, 0, 1);          // global_prefetch_b8
    #pragma unroll
    for (int q = 0; q < F_/4; ++q) {
      const float4 v = ((const float4*)xr)[q];
      row[q*4+0] = (_Float16)v.x; row[q*4+1] = (_Float16)v.y;
      row[q*4+2] = (_Float16)v.z; row[q*4+3] = (_Float16)v.w;
    }
    row[F_+0] = (_Float16)p0;  row[F_+1] = (_Float16)an1;
    row[F_+2] = (_Float16)an2; row[F_+3] = (_Float16)an3;
    #pragma unroll
    for (int q = F_+4; q < KP_; ++q) row[q] = (_Float16)0.f;
    flagL[tid] = (d2 <= R2_) ? 1.f : 0.f;

    if (lane == 0) {
      outPos [(size_t)g*3+0] = pix; outPos [(size_t)g*3+1] = piy; outPos [(size_t)g*3+2] = piz;
      outNorm[(size_t)g*3+0] = nix; outNorm[(size_t)g*3+1] = niy; outNorm[(size_t)g*3+2] = niz;
    }
  }
  __syncthreads();

  const int rowbase = wid * K_;

  // ---- GEMM1: [32 x 96] x [96 x 64] per wave, f16 WMMA, f32 acc ----
  v16h a1f[2][3];
  #pragma unroll
  for (int rt = 0; rt < 2; ++rt)
    #pragma unroll
    for (int ks = 0; ks < 3; ++ks)
      a1f[rt][ks] = load_fragA(msgL, rowbase + rt*16, KP_, ks*32, lane);

  #pragma unroll
  for (int ct = 0; ct < H1_/16; ++ct) {
    v8f acc0 = {}; v8f acc1 = {};
    #pragma unroll
    for (int ks = 0; ks < 3; ++ks) {
      const v16h bf = load_fragB(wt1, ct*16, KP_, ks*32, lane);
      acc0 = __builtin_amdgcn_wmma_f32_16x16x32_f16(false, a1f[0][ks], false, bf,
                                                    (short)0, acc0, false, false);
      acc1 = __builtin_amdgcn_wmma_f32_16x16x32_f16(false, a1f[1][ks], false, bf,
                                                    (short)0, acc1, false, false);
    }
    const float bb = b1L[ct*16 + (lane & 15)];
    const int ro = (lane & 16) ? 8 : 0;
    const int cidx = ct*16 + (lane & 15);
    #pragma unroll
    for (int jj = 0; jj < 8; ++jj) {
      const float v0 = fmaxf(acc0[jj] + bb, 0.f);
      const float v1 = fmaxf(acc1[jj] + bb, 0.f);
      hL[(rowbase +      jj + ro) * H1_ + cidx] = (_Float16)v0;
      hL[(rowbase + 16 + jj + ro) * H1_ + cidx] = (_Float16)v1;
    }
  }
  // hL rows are wave-private: per-wave DS ordering covers the RAW, no barrier.

  // ---- per-lane row-valid flags for the C-tile layout ----
  float fl[2][8];
  const int roff = (lane & 16) ? 8 : 0;
  #pragma unroll
  for (int rt = 0; rt < 2; ++rt)
    #pragma unroll
    for (int jj = 0; jj < 8; ++jj)
      fl[rt][jj] = flagL[rowbase + rt*16 + jj + roff];
  const float anyV = flagL[rowbase];       // k=0 is the nearest neighbor

  // ---- GEMM2: [32 x 64] x [64 x 128] per wave + masked max over K ----
  v16h a2f[2][2];
  #pragma unroll
  for (int rt = 0; rt < 2; ++rt)
    #pragma unroll
    for (int ks = 0; ks < 2; ++ks)
      a2f[rt][ks] = load_fragA(hL, rowbase + rt*16, H1_, ks*32, lane);

  #pragma unroll
  for (int ct = 0; ct < H2_/16; ++ct) {
    v8f acc0 = {}; v8f acc1 = {};
    #pragma unroll
    for (int ks = 0; ks < 2; ++ks) {
      const v16h bf = load_fragB(wt2, ct*16, H1_, ks*32, lane);
      acc0 = __builtin_amdgcn_wmma_f32_16x16x32_f16(false, a2f[0][ks], false, bf,
                                                    (short)0, acc0, false, false);
      acc1 = __builtin_amdgcn_wmma_f32_16x16x32_f16(false, a2f[1][ks], false, bf,
                                                    (short)0, acc1, false, false);
    }
    const float bb = b2L[ct*16 + (lane & 15)];
    float mloc = -3.4e38f;
    #pragma unroll
    for (int jj = 0; jj < 8; ++jj) {
      float v0 = fmaxf(acc0[jj] + bb, 0.f);
      float v1 = fmaxf(acc1[jj] + bb, 0.f);
      v0 = fl[0][jj] > 0.f ? v0 : -3.4e38f;
      v1 = fl[1][jj] > 0.f ? v1 : -3.4e38f;
      mloc = fmaxf(mloc, fmaxf(v0, v1));
    }
    mloc = fmaxf(mloc, __shfl_xor(mloc, 16, 32));   // combine lane halves (M 0-7 / 8-15)
    if (lane < 16)
      out[(size_t)g*H2_ + ct*16 + lane] = (anyV > 0.f) ? mloc : 0.f;
  }
}

extern "C" void kernel_launch(void* const* d_in, const int* in_sizes, int n_in,
                              void* d_out, int out_size, void* d_ws, size_t ws_size,
                              hipStream_t stream) {
  (void)in_sizes; (void)n_in; (void)out_size; (void)ws_size;
  const float* x    = (const float*)d_in[0];
  const float* pos  = (const float*)d_in[1];
  const float* norm = (const float*)d_in[2];
  const float* W1   = (const float*)d_in[3];
  const float* b1   = (const float*)d_in[4];
  const float* W2   = (const float*)d_in[5];
  const float* b2   = (const float*)d_in[6];

  float* out     = (float*)d_out;                       // [B, M, H2]
  float* outPos  = out + (size_t)B_*M_*H2_;             // [B, M, 3]
  float* outNorm = outPos + (size_t)B_*M_*3;            // [B, M, 3]

  int* idxArr = (int*)d_ws;                             // B*M
  int* nbrArr = idxArr + (size_t)B_*M_;                 // B*M*K

  const size_t fpsLds  = (size_t)3*N_*4 + 32*4 + 32*4 + 16;
  const size_t topkLds = (size_t)8*N_*4;
  const size_t mlpLds  = (size_t)(256*KP_ + H1_*KP_ + H2_*H1_ + 256*H1_) * 2
                       + (size_t)(H1_ + H2_ + 256) * 4;

  fps_kernel <<<B_,            1024, fpsLds,  stream>>>(pos, idxArr);
  topk_kernel<<<(B_*M_)/8,      256, topkLds, stream>>>(pos, idxArr, nbrArr);
  ppf_mlp_kernel<<<(B_*M_)/8,   256, mlpLds,  stream>>>(x, pos, norm, W1, b1, W2, b2,
                                                        idxArr, nbrArr,
                                                        out, outPos, outNorm);
}
